// Encoder_layer_11441792876634
// MI455X (gfx1250) — matrix operand order repaired and checked by reading, not compiled, not run on hardware
//
#include <hip/hip_runtime.h>
#include <hip/hip_bf16.h>

typedef __bf16 bf16_t;
typedef __attribute__((ext_vector_type(16))) __bf16 v16bf;
typedef __attribute__((ext_vector_type(8)))  __bf16 v8bf;
typedef __attribute__((ext_vector_type(8)))  float   v8f;

// ---------------------------------------------------------------------------
// WMMA helpers (CDNA5: V_WMMA_F32_16X16X32_BF16, wave32)
// ---------------------------------------------------------------------------
__device__ __forceinline__ v8f wmma_bf16(v16bf a, v16bf b, v8f c) {
  return __builtin_amdgcn_wmma_f32_16x16x32_bf16(
      /*neg_a=*/false, a, /*neg_b=*/false, b,
      /*c_mod=*/(short)0, c, /*reuse_a=*/false, /*reuse_b=*/false);
}

// A-fragment (16x32, MxK): per-lane row M=lane%16; lanes 0-15 hold K=0..15,
// lanes 16-31 hold K=16..31 (contiguous). Caller passes p = row_base + koffA (0 or 16).
__device__ __forceinline__ v16bf ld_a_frag(const bf16_t* p) {
  v8bf lo = *(const v8bf*)(p);
  v8bf hi = *(const v8bf*)(p + 8);
  return __builtin_shufflevector(lo, hi, 0,1,2,3,4,5,6,7,8,9,10,11,12,13,14,15);
}

// B-fragment (32x16, KxN): per-lane column N=lane%16; lanes 0-15 hold K=0..15,
// lanes 16-31 hold K=16..31 (contiguous). Caller passes p = col_base + koffB (0 or 16).
__device__ __forceinline__ v16bf ld_b_frag(const bf16_t* p) {
  v8bf lo = *(const v8bf*)(p);
  v8bf hi = *(const v8bf*)(p + 8);
  return __builtin_shufflevector(lo, hi, 0,1,2,3,4,5,6,7,8,9,10,11,12,13,14,15);
}

// ---------------------------------------------------------------------------
// CDNA5 async global->LDS DMA (no data VGPRs, tracked by ASYNCcnt)
// ---------------------------------------------------------------------------
__device__ __forceinline__ void async_load_b128(unsigned lds_addr, const void* gaddr) {
  asm volatile("global_load_async_to_lds_b128 %0, %1, off"
               :: "v"(lds_addr), "v"(gaddr) : "memory");
}
__device__ __forceinline__ void wait_asynccnt0() {
  asm volatile("s_wait_asynccnt 0x0" ::: "memory");
}

// ---------------------------------------------------------------------------
// fp32 -> bf16 convert (x)
// ---------------------------------------------------------------------------
__global__ __launch_bounds__(256) void cvt_bf16_kernel(
    const float* __restrict__ in, bf16_t* __restrict__ out) {
  long i = ((long)blockIdx.x * 256 + threadIdx.x) * 4;
  float4 f = *(const float4*)(in + i);
  out[i + 0] = (bf16_t)f.x;
  out[i + 1] = (bf16_t)f.y;
  out[i + 2] = (bf16_t)f.z;
  out[i + 3] = (bf16_t)f.w;
}

// ---------------------------------------------------------------------------
// Transpose + convert weights: in fp32 [K,N] -> out bf16 [N,K]
// ---------------------------------------------------------------------------
__global__ __launch_bounds__(256) void transpose_bf16_kernel(
    const float* __restrict__ in, bf16_t* __restrict__ out, int K, int N) {
  __shared__ float t[32][33];
  int n0 = blockIdx.x * 32, k0 = blockIdx.y * 32;
#pragma unroll
  for (int i = 0; i < 4; i++) {
    int kk = threadIdx.y + i * 8;
    t[kk][threadIdx.x] = in[(long)(k0 + kk) * N + n0 + threadIdx.x];
  }
  __syncthreads();
#pragma unroll
  for (int i = 0; i < 4; i++) {
    int nn = threadIdx.y + i * 8;
    out[(long)(n0 + nn) * K + k0 + threadIdx.x] = (bf16_t)t[threadIdx.x][nn];
  }
}

// ---------------------------------------------------------------------------
// Generic bf16 WMMA GEMM:  C[M,N] = A[M,K] @ Bt[N,K]^T, epilogue fusions.
// Block tile 128x128, K-stage 32, 8 waves (wave: 64x32 tile).
// Double-buffered LDS filled by GLOBAL_LOAD_ASYNC_TO_LDS_B128 (zero data
// VGPRs), one s_wait_asynccnt + one barrier per stage.
// ---------------------------------------------------------------------------
#define BM 128
#define BN 128
#define BKS 32
#define LDK 40  // padded LDS row stride (bf16 elems), 80B = 16B aligned

template <bool OUT_BF16, bool RELU, bool HAS_RES>
__global__ __launch_bounds__(256) void gemm_kernel(
    const bf16_t* __restrict__ A, const bf16_t* __restrict__ Bt,
    const float* __restrict__ bias, const float* __restrict__ res,
    float scale, void* __restrict__ out, int M, int N, int K) {
  __shared__ bf16_t As[2][BM * LDK];
  __shared__ bf16_t Bs[2][BN * LDK];
  const int tid   = threadIdx.x;
  const int lane  = tid & 31;
  const int wid   = tid >> 5;
  const int lrow  = lane & 15;
  const int koffA = (lane >> 4) << 4;
  const int koffB = (lane >> 4) << 4;
  const int waveM = (wid & 1) * 64;
  const int waveN = (wid >> 1) * 32;
  const long bMr = (long)blockIdx.y * BM;
  const long bNr = (long)blockIdx.x * BN;

  // Per-thread stage-load coordinates: 128 rows x 32 cols, 16B chunks.
  // 512 chunks per matrix, 256 threads -> 2 chunks each.
  int srow[2], scol[2];
#pragma unroll
  for (int j = 0; j < 2; j++) {
    int idx = tid + j * 256;
    srow[j] = idx >> 2;
    scol[j] = (idx & 3) * 8;
  }
  const unsigned asBase = (unsigned)(size_t)&As[0][0];
  const unsigned bsBase = (unsigned)(size_t)&Bs[0][0];
  const unsigned bufStride = BM * LDK * 2;  // bytes per buffer

  // Prologue: DMA stage 0 into buffer 0
#pragma unroll
  for (int j = 0; j < 2; j++) {
    unsigned loff = (unsigned)(srow[j] * LDK + scol[j]) * 2;
    async_load_b128(asBase + loff, (const void*)(A  + (bMr + srow[j]) * K + scol[j]));
    async_load_b128(bsBase + loff, (const void*)(Bt + (bNr + srow[j]) * K + scol[j]));
  }

  v8f acc[4][2];
#pragma unroll
  for (int i = 0; i < 4; i++)
#pragma unroll
    for (int j = 0; j < 2; j++)
#pragma unroll
      for (int r = 0; r < 8; r++) acc[i][j][r] = 0.0f;

  int buf = 0;
  for (int k0 = 0; k0 < K; k0 += BKS) {
    wait_asynccnt0();     // our DMA for current buffer done
    __syncthreads();      // everyone's DMA done; prev compute done (buffer safe)
    if (k0 + BKS < K) {   // DMA next stage into the other buffer
      unsigned nb = (unsigned)(buf ^ 1) * bufStride;
#pragma unroll
      for (int j = 0; j < 2; j++) {
        unsigned loff = nb + (unsigned)(srow[j] * LDK + scol[j]) * 2;
        async_load_b128(asBase + loff,
                        (const void*)(A  + (bMr + srow[j]) * K + k0 + BKS + scol[j]));
        async_load_b128(bsBase + loff,
                        (const void*)(Bt + (bNr + srow[j]) * K + k0 + BKS + scol[j]));
      }
    }
    // Compute current stage: 8 WMMAs
    v16bf af[4];
#pragma unroll
    for (int mf = 0; mf < 4; mf++)
      af[mf] = ld_a_frag(&As[buf][(waveM + mf * 16 + lrow) * LDK + koffA]);
#pragma unroll
    for (int nf = 0; nf < 2; nf++) {
      v16bf bfr = ld_b_frag(&Bs[buf][(waveN + nf * 16 + lrow) * LDK + koffB]);
#pragma unroll
      for (int mf = 0; mf < 4; mf++)
        acc[mf][nf] = wmma_bf16(af[mf], bfr, acc[mf][nf]);
    }
    buf ^= 1;
  }

  // Epilogue. C-layout: lanes 0-15 -> N=lane, M=row r; lanes 16-31 -> M=8+r.
  const int moff = (lane >> 4) << 3;
#pragma unroll
  for (int mf = 0; mf < 4; mf++) {
#pragma unroll
    for (int nf = 0; nf < 2; nf++) {
      long gn = bNr + waveN + nf * 16 + lrow;
      float bv = bias[gn];
#pragma unroll
      for (int r = 0; r < 8; r++) {
        long gm = bMr + waveM + mf * 16 + moff + r;
        float v = (acc[mf][nf][r] + bv) * scale;
        if (HAS_RES) v += res[gm * N + gn];
        if (RELU) v = v > 0.0f ? v : 0.0f;
        if (OUT_BF16) ((bf16_t*)out)[gm * N + gn] = (bf16_t)v;
        else          ((float*)out)[gm * N + gn] = v;
      }
    }
  }
}

// ---------------------------------------------------------------------------
// Flash attention (dilated groups). Computes S^T = K @ Q^T, online softmax
// over keys, ctx^T += V^T @ P^T. Q pre-scaled by 1/sqrt(dh) at QKV GEMM.
// Grid: (qtile=8, head=16, group=8). Block: 8 waves; wave owns 16 queries.
// Spill-free round-1 structure: direct loads at stage top, exp in place.
// ---------------------------------------------------------------------------
#define AT_SC 128   // keys per LDS stage
#define KS_LD 72    // Ks row stride (bf16 elems), 144B
#define VS_LD 136   // Vs row stride (bf16 elems), 272B

__global__ __launch_bounds__(256) void attn_kernel(
    const bf16_t* __restrict__ Qb, const bf16_t* __restrict__ Kb,
    const bf16_t* __restrict__ Vb, bf16_t* __restrict__ ctxb) {
  __shared__ bf16_t Ks[AT_SC * KS_LD];   // [key][dh]
  __shared__ bf16_t Vs[64 * VS_LD];      // [dh][key] (transposed V)
  const int tid    = threadIdx.x;
  const int lane   = tid & 31;
  const int wid    = tid >> 5;
  const int lrow   = lane & 15;
  const int hihalf = lane >> 4;
  const int koffA  = hihalf << 4;
  const int koffB  = hihalf << 4;
  const int head   = blockIdx.y;
  const int grp    = blockIdx.z;
  const int bb = grp >> 1, rem = grp & 1;
  const long hoff = (long)head * 64;

  // This lane's query column (N dim of S^T)
  const int qloc = blockIdx.x * 128 + wid * 16 + lrow;
  const long tq  = (long)bb * 2048 + qloc * 2 + rem;
  const bf16_t* qp = Qb + tq * 1024 + hoff;
  v16bf qf0 = ld_b_frag(qp + koffB);        // dh 0..31
  v16bf qf1 = ld_b_frag(qp + 32 + koffB);   // dh 32..63

  v8f accc[4];  // ctx^T: 4 feature frags x 16 queries
#pragma unroll
  for (int i = 0; i < 4; i++)
#pragma unroll
    for (int r = 0; r < 8; r++) accc[i][r] = 0.0f;
  float mrun = -3.0e30f, lrun = 0.0f;

  for (int ks = 0; ks < 1024 / AT_SC; ks++) {
    __syncthreads();
#pragma unroll
    for (int j = 0; j < 4; j++) {
      int idx = tid + j * 256;
      int row = idx >> 3;
      int c   = (idx & 7) * 8;
      long tk = (long)bb * 2048 + (ks * AT_SC + row) * 2 + rem;
      *(uint4*)&Ks[row * KS_LD + c] = *(const uint4*)(Kb + tk * 1024 + hoff + c);
      v8bf vv = *(const v8bf*)(Vb + tk * 1024 + hoff + c);
#pragma unroll
      for (int f = 0; f < 8; f++) Vs[(c + f) * VS_LD + row] = vv[f];
    }
    __syncthreads();

    for (int kc = 0; kc < AT_SC / 32; kc++) {
      const int kb = kc * 32;
      v8f s0, s1;
#pragma unroll
      for (int r = 0; r < 8; r++) { s0[r] = 0.0f; s1[r] = 0.0f; }
      s0 = wmma_bf16(ld_a_frag(&Ks[(kb + lrow) * KS_LD + koffA]),        qf0, s0);
      s0 = wmma_bf16(ld_a_frag(&Ks[(kb + lrow) * KS_LD + 32 + koffA]),   qf1, s0);
      s1 = wmma_bf16(ld_a_frag(&Ks[(kb + 16 + lrow) * KS_LD + koffA]),      qf0, s1);
      s1 = wmma_bf16(ld_a_frag(&Ks[(kb + 16 + lrow) * KS_LD + 32 + koffA]), qf1, s1);

      // Online softmax over this 32-key chunk (per query = per lane-pair)
      float mloc = -3.0e30f;
#pragma unroll
      for (int r = 0; r < 8; r++) mloc = fmaxf(mloc, fmaxf(s0[r], s1[r]));
      mloc = fmaxf(mloc, __shfl_xor(mloc, 16, 32));
      float mnew  = fmaxf(mrun, mloc);
      float alpha = __expf(mrun - mnew);
      // exp in place over score registers (P replaces S)
      float ssum = 0.0f;
#pragma unroll
      for (int r = 0; r < 8; r++) {
        s0[r] = __expf(s0[r] - mnew);
        s1[r] = __expf(s1[r] - mnew);
        ssum += s0[r] + s1[r];
      }
      ssum += __shfl_xor(ssum, 16, 32);
      lrun = lrun * alpha + ssum;
      mrun = mnew;

      // Build P^T B-fragment: lanes 0-15 need keys 0..15 (own p0 + partner p0),
      // lanes 16-31 need keys 16..31 (partner p1 + own p1). One shuffle per r.
      v16bf pb;
#pragma unroll
      for (int r = 0; r < 8; r++) {
        float sendv = hihalf ? s0[r] : s1[r];
        float recv  = __shfl_xor(sendv, 16, 32);
        pb[r]     = (bf16_t)(hihalf ? recv  : s0[r]);
        pb[8 + r] = (bf16_t)(hihalf ? s1[r] : recv);
      }

#pragma unroll
      for (int ff = 0; ff < 4; ff++)
#pragma unroll
        for (int r = 0; r < 8; r++) accc[ff][r] *= alpha;
#pragma unroll
      for (int ff = 0; ff < 4; ff++)
        accc[ff] = wmma_bf16(
            ld_a_frag(&Vs[(ff * 16 + lrow) * VS_LD + kb + koffA]), pb, accc[ff]);
    }
  }

  // ctx^T C-layout: lane holds query=lrow, features ff*16 + (hihalf?8:0) + r
  // -> 8 contiguous features per lane: direct coalesced bf16 store.
  float invl = 1.0f / lrun;
  const int foff = hihalf << 3;
#pragma unroll
  for (int ff = 0; ff < 4; ff++) {
    v8bf o;
#pragma unroll
    for (int r = 0; r < 8; r++) o[r] = (bf16_t)(accc[ff][r] * invl);
    *(v8bf*)(ctxb + tq * 1024 + hoff + ff * 16 + foff) = o;
  }
}

// ---------------------------------------------------------------------------
// LayerNorm over 1024 features; optional bf16 mirror for next GEMM A-operand.
// ---------------------------------------------------------------------------
__global__ __launch_bounds__(256) void ln_kernel(
    const float* __restrict__ in, const float* __restrict__ gam,
    const float* __restrict__ bet, float* __restrict__ out32,
    bf16_t* __restrict__ out16) {
  __shared__ float red[8];
  const int row = blockIdx.x;
  const int tid = threadIdx.x;
  const float* p = in + (long)row * 1024;
  float v[4];
  float s = 0.0f;
#pragma unroll
  for (int i = 0; i < 4; i++) { v[i] = p[tid + i * 256]; s += v[i]; }
#pragma unroll
  for (int o = 16; o > 0; o >>= 1) s += __shfl_xor(s, o, 32);
  if ((tid & 31) == 0) red[tid >> 5] = s;
  __syncthreads();
  float tot = 0.0f;
#pragma unroll
  for (int i = 0; i < 8; i++) tot += red[i];
  float mu = tot * (1.0f / 1024.0f);
  __syncthreads();
  float var = 0.0f;
#pragma unroll
  for (int i = 0; i < 4; i++) { float d = v[i] - mu; var += d * d; }
#pragma unroll
  for (int o = 16; o > 0; o >>= 1) var += __shfl_xor(var, o, 32);
  if ((tid & 31) == 0) red[tid >> 5] = var;
  __syncthreads();
  float vt = 0.0f;
#pragma unroll
  for (int i = 0; i < 8; i++) vt += red[i];
  float rstd = rsqrtf(vt * (1.0f / 1024.0f) + 1e-5f);
#pragma unroll
  for (int i = 0; i < 4; i++) {
    int c = tid + i * 256;
    float o = (v[i] - mu) * rstd * gam[c] + bet[c];
    out32[(long)row * 1024 + c] = o;
    if (out16) out16[(long)row * 1024 + c] = (bf16_t)o;
  }
}

// ---------------------------------------------------------------------------
// Host launcher
// ---------------------------------------------------------------------------
extern "C" void kernel_launch(void* const* d_in, const int* in_sizes, int n_in,
                              void* d_out, int out_size, void* d_ws, size_t ws_size,
                              hipStream_t stream) {
  (void)in_sizes; (void)n_in; (void)out_size; (void)ws_size;
  const float* x     = (const float*)d_in[0];
  const float* Wq    = (const float*)d_in[1];
  const float* bq    = (const float*)d_in[2];
  const float* Wk    = (const float*)d_in[3];
  const float* bk    = (const float*)d_in[4];
  const float* Wv    = (const float*)d_in[5];
  const float* bv    = (const float*)d_in[6];
  const float* Wo    = (const float*)d_in[7];
  const float* bo    = (const float*)d_in[8];
  const float* ln1_g = (const float*)d_in[9];
  const float* ln1_b = (const float*)d_in[10];
  const float* W1    = (const float*)d_in[11];
  const float* b1    = (const float*)d_in[12];
  const float* W2    = (const float*)d_in[13];
  const float* b2    = (const float*)d_in[14];
  const float* ln2_g = (const float*)d_in[15];
  const float* ln2_b = (const float*)d_in[16];

  char* ws = (char*)d_ws;
  const size_t MB = 1024 * 1024;
  bf16_t* WqT   = (bf16_t*)(ws + 0 * MB);
  bf16_t* WkT   = (bf16_t*)(ws + 2 * MB);
  bf16_t* WvT   = (bf16_t*)(ws + 4 * MB);
  bf16_t* WoT   = (bf16_t*)(ws + 6 * MB);
  bf16_t* W1T   = (bf16_t*)(ws + 8 * MB);
  bf16_t* W2T   = (bf16_t*)(ws + 16 * MB);
  bf16_t* xb    = (bf16_t*)(ws + 24 * MB);
  bf16_t* Qb    = (bf16_t*)(ws + 40 * MB);
  bf16_t* Kb    = (bf16_t*)(ws + 56 * MB);
  bf16_t* Vb    = (bf16_t*)(ws + 72 * MB);
  bf16_t* ctxb  = (bf16_t*)(ws + 88 * MB);
  bf16_t* hb    = (bf16_t*)(ws + 40 * MB);   // reuse Qb..ctxb (64MB)
  float*  atted = (float*)(ws + 104 * MB);
  float*  y2    = (float*)(ws + 104 * MB);   // reuse atted
  float*  finf  = (float*)(ws + 136 * MB);
  bf16_t* finb  = (bf16_t*)(ws + 168 * MB);  // total 184MB

  dim3 tb(32, 8);
  cvt_bf16_kernel<<<8192, 256, 0, stream>>>(x, xb);
  transpose_bf16_kernel<<<dim3(32, 32), tb, 0, stream>>>(Wq, WqT, 1024, 1024);
  transpose_bf16_kernel<<<dim3(32, 32), tb, 0, stream>>>(Wk, WkT, 1024, 1024);
  transpose_bf16_kernel<<<dim3(32, 32), tb, 0, stream>>>(Wv, WvT, 1024, 1024);
  transpose_bf16_kernel<<<dim3(32, 32), tb, 0, stream>>>(Wo, WoT, 1024, 1024);
  transpose_bf16_kernel<<<dim3(128, 32), tb, 0, stream>>>(W1, W1T, 1024, 4096);
  transpose_bf16_kernel<<<dim3(32, 128), tb, 0, stream>>>(W2, W2T, 4096, 1024);

  dim3 gD(1024 / 128, 8192 / 128);  // N=1024 GEMMs
  // Q pre-scaled by 1/sqrt(64)
  gemm_kernel<true,  false, false><<<gD, 256, 0, stream>>>(xb, WqT, bq, nullptr, 0.125f, Qb, 8192, 1024, 1024);
  gemm_kernel<true,  false, false><<<gD, 256, 0, stream>>>(xb, WkT, bk, nullptr, 1.0f,   Kb, 8192, 1024, 1024);
  gemm_kernel<true,  false, false><<<gD, 256, 0, stream>>>(xb, WvT, bv, nullptr, 1.0f,   Vb, 8192, 1024, 1024);

  attn_kernel<<<dim3(8, 16, 8), 256, 0, stream>>>(Qb, Kb, Vb, ctxb);

  gemm_kernel<false, false, true ><<<gD, 256, 0, stream>>>(ctxb, WoT, bo, x, 1.0f, atted, 8192, 1024, 1024);
  ln_kernel<<<8192, 256, 0, stream>>>(atted, ln1_g, ln1_b, finf, finb);

  gemm_kernel<true,  true,  false><<<dim3(4096 / 128, 8192 / 128), 256, 0, stream>>>(finb, W1T, b1, nullptr, 1.0f, hb, 8192, 4096, 1024);
  gemm_kernel<false, false, true ><<<gD, 256, 0, stream>>>(hb, W2T, b2, finf, 1.0f, y2, 8192, 1024, 4096);
  ln_kernel<<<8192, 256, 0, stream>>>(y2, ln2_g, ln2_b, (float*)d_out, nullptr);
}
